// _DCN_Module_34033320854095
// MI455X (gfx1250) — matrix operand order repaired and checked by reading, not compiled, not run on hardware
//
#include <hip/hip_runtime.h>

#define NN    65536
#define DD    256
#define KK    512
#define ROWS  128      // rows per block (8 waves x 16 rows)
#define CHUNK 64       // center columns staged in LDS per chunk
#define BPAD  264      // padded row stride (bf16 elems) -> conflict-free ds_load

typedef __attribute__((ext_vector_type(16))) __bf16 v16bf_t;
typedef __attribute__((ext_vector_type(8)))  float  v8f_t;
typedef unsigned int u32x4 __attribute__((ext_vector_type(4)));
typedef int          i32x4 __attribute__((ext_vector_type(4)));
typedef int          i32x8 __attribute__((ext_vector_type(8)));

#if defined(__has_builtin)
#  if __has_builtin(__builtin_amdgcn_tensor_load_to_lds) && \
      __has_builtin(__builtin_amdgcn_s_wait_tensorcnt)
#    define HAVE_TDM 1
#  endif
#endif
#ifndef HAVE_TDM
#  define HAVE_TDM 0
#endif

static __device__ __forceinline__ unsigned short f2bf(float f) {
  unsigned u = __builtin_bit_cast(unsigned, f);
  u = (u + 0x7FFFu + ((u >> 16) & 1u)) >> 16;   // round-to-nearest-even
  return (unsigned short)u;
}

// ---------------------------------------------------------------- zero ws ---
__global__ __launch_bounds__(256) void dcn_zero(float* p, int n) {
  int i = blockIdx.x * 256 + threadIdx.x;
  if (i < n) p[i] = 0.0f;
}

// ------------------------- per-center ||b||^2 + bf16 conversion of centers --
__global__ __launch_bounds__(256) void dcn_prep(const float* __restrict__ cen,
                                                float* __restrict__ b2,
                                                unsigned short* __restrict__ cenBf) {
  const int wave = threadIdx.x >> 5;
  const int lane = threadIdx.x & 31;
  const int row  = blockIdx.x * 8 + wave;
  const float* p = cen + (size_t)row * DD;
  unsigned short* q = cenBf + (size_t)row * DD;
  float s = 0.0f;
#pragma unroll
  for (int j = 0; j < DD / 32; ++j) {
    float v = p[lane + j * 32];
    q[lane + j * 32] = f2bf(v);
    s += v * v;
  }
#pragma unroll
  for (int off = 16; off > 0; off >>= 1) s += __shfl_xor(s, off);
  if (lane == 0) b2[row] = s;
}

// -------------------------------------------- fused WMMA distance + argmin --
__global__ __launch_bounds__(256)
void dcn_main(const float* __restrict__ emb, const float* __restrict__ cen,
              const unsigned short* __restrict__ cenBf,
              const float* __restrict__ b2, float* __restrict__ labelsOut,
              float* __restrict__ segSum, int* __restrict__ segCnt,
              float* __restrict__ lossAcc) {
  __shared__ unsigned short sB[CHUNK * BPAD];   // bf16 center chunk, padded
  __shared__ int   sLab[ROWS];
  __shared__ float sLoss;

  const int tid  = threadIdx.x;
  const int wave = tid >> 5;
  const int lane = tid & 31;
  const int half = lane >> 4;          // 0: rows use K 0..7/16..23 ; 1: 8..15/24..31
  const int l16  = lane & 15;
  const int blockRow = blockIdx.x * ROWS;
  const int rowBase  = blockRow + wave * 16;

  if (tid == 0) sLoss = 0.0f;

  // ---- load A panel (16 rows x 256) as bf16 fragments in documented layout
  union AF { v16bf_t v; unsigned short u[16]; };
  AF afr[8];
  float a2part = 0.0f;
#pragma unroll
  for (int s = 0; s < 8; ++s) {
    const float* src = emb + (size_t)(rowBase + l16) * DD + s * 32 + half * 8;
    float4 p0 = *(const float4*)(src + 0);
    float4 p1 = *(const float4*)(src + 4);
    float4 q0 = *(const float4*)(src + 16);
    float4 q1 = *(const float4*)(src + 20);
    float pv[8] = {p0.x, p0.y, p0.z, p0.w, p1.x, p1.y, p1.z, p1.w};
    float qv[8] = {q0.x, q0.y, q0.z, q0.w, q1.x, q1.y, q1.z, q1.w};
#pragma unroll
    for (int e = 0; e < 8; ++e) {
      afr[s].u[e]     = f2bf(pv[e]);   // K = half*8 + e
      afr[s].u[e + 8] = f2bf(qv[e]);   // K = 16 + half*8 + e
      a2part += pv[e] * pv[e] + qv[e] * qv[e];
    }
  }
  // lane L and L^16 hold complementary K halves of row (lane&15)
  float a2full = a2part + __shfl_xor(a2part, 16);
  float a2r[8];
#pragma unroll
  for (int r = 0; r < 8; ++r) a2r[r] = __shfl(a2full, half * 8 + r);

  float bestD[8];
  int   bestI[8];
#pragma unroll
  for (int r = 0; r < 8; ++r) { bestD[r] = 3.4e38f; bestI[r] = 0; }

  // ---- sweep all K centers in LDS-staged chunks
  for (int chunk = 0; chunk < KK / CHUNK; ++chunk) {
    if (chunk + 1 < KK / CHUNK)
      __builtin_prefetch(cenBf + (size_t)((chunk + 1) * CHUNK) * DD + tid * 8, 0, 1);
    __syncthreads();
#if HAVE_TDM
    // Tensor Data Mover: 2D bf16 tile copy [64 cols x 256 d] with HW LDS pad
    // (pad_interval=6 -> every 128 DWORDs, pad_amount=3 -> 4 DWORDs => BPAD=264)
    if (tid == 0) {
      unsigned long long ga =
          (unsigned long long)(uintptr_t)(cenBf + (size_t)chunk * CHUNK * DD);
      unsigned ldsOff = (unsigned)(uintptr_t)(&sB[0]);  // low 32 bits = LDS offset
      u32x4 g0;
      g0.x = 1u;                                             // count=1, load desc
      g0.y = ldsOff;                                         // lds_addr [63:32]
      g0.z = (unsigned)ga;                                   // global_addr lo
      g0.w = (unsigned)((ga >> 32) & 0x1FFFFFFull) | (2u << 30); // addr hi | type=2
      i32x8 g1;
      g1[0] = (int)((1u << 16)    // data_size = 2 bytes
                  | (1u << 20)    // pad_enable
                  | (6u << 22)    // pad_interval: 2^6 * 8B = 128 DWORDs
                  | (3u << 25));  // pad_amount: 4 DWORDs
      g1[1] = (int)(256u << 16);  // tensor_dim0[15:0] in bits [63:48]
      g1[2] = (int)(512u << 16);  // tensor_dim1[15:0] in bits [95:80]
      g1[3] = (int)(256u << 16);  // tile_dim0 in bits [127:112]
      g1[4] = 64;                 // tile_dim1 in bits [143:128]
      g1[5] = 256;                // tensor_dim0_stride lo
      g1[6] = 0;
      g1[7] = 0;
      i32x4 z4 = {0, 0, 0, 0};
#if __clang_major__ >= 23
      i32x8 z8 = {0, 0, 0, 0, 0, 0, 0, 0};
      __builtin_amdgcn_tensor_load_to_lds(g0, g1, z4, z4, z8, 0);
#else
      __builtin_amdgcn_tensor_load_to_lds(g0, g1, z4, z4, 0);
#endif
      __builtin_amdgcn_s_wait_tensorcnt(0);
    }
#else
    for (int idx4 = tid; idx4 < CHUNK * DD / 4; idx4 += 256) {
      int col = idx4 >> 6;                 // DD/4 = 64 quads per column
      int dq  = idx4 & 63;
      float4 v = *(const float4*)(cen + (size_t)(chunk * CHUNK + col) * DD + dq * 4);
      unsigned short* dst = &sB[col * BPAD + dq * 4];
      dst[0] = f2bf(v.x); dst[1] = f2bf(v.y);
      dst[2] = f2bf(v.z); dst[3] = f2bf(v.w);
    }
#endif
    __syncthreads();

#pragma unroll
    for (int ct = 0; ct < CHUNK / 16; ++ct) {
      const int colBase = chunk * CHUNK + ct * 16;
      v8f_t acc = {};
#pragma unroll
      for (int s = 0; s < 8; ++s) {
        // B frag: N = l16, element e -> K = half*8 + e (e<8), 16 + half*8 + (e-8) (e>=8)
        union BF { v16bf_t v; uint4 q[2]; } bf_;
        const uint4* bp =
            (const uint4*)&sB[(ct * 16 + l16) * BPAD + s * 32 + half * 8];
        bf_.q[0] = bp[0];
        bf_.q[1] = bp[2];
        acc = __builtin_amdgcn_wmma_f32_16x16x32_bf16(
            false, afr[s].v, false, bf_.v, (short)0, acc, false, false);
      }
      float b2c = b2[colBase + l16];
      int   idx = colBase + l16;
#pragma unroll
      for (int r = 0; r < 8; ++r) {
        float dist = a2r[r] + b2c - 2.0f * acc[r];
        if (dist < bestD[r]) { bestD[r] = dist; bestI[r] = idx; }
      }
    }
  }

  // ---- cross-lane argmin within the 16-lane half (those share the 8 rows)
#pragma unroll
  for (int off = 1; off < 16; off <<= 1) {
#pragma unroll
    for (int r = 0; r < 8; ++r) {
      float od = __shfl_xor(bestD[r], off);
      int   oi = __shfl_xor(bestI[r], off);
      if (od < bestD[r] || (od == bestD[r] && oi < bestI[r])) {
        bestD[r] = od; bestI[r] = oi;
      }
    }
  }
  if (l16 == 0) {                       // lanes 0 and 16 of each wave
    float ls = 0.0f;
#pragma unroll
    for (int r = 0; r < 8; ++r) {
      int m = half * 8 + r;
      labelsOut[rowBase + m] = (float)bestI[r];
      sLab[wave * 16 + m]    = bestI[r];
      ls += bestD[r];
    }
    atomicAdd(&sLoss, ls);
  }
  __syncthreads();
  if (tid == 0) atomicAdd(lossAcc, sLoss);

  // ---- segment sums: thread = d, loop rows
  for (int r = 0; r < ROWS; ++r) {
    int lbl = sLab[r];
    float v = emb[(size_t)(blockRow + r) * DD + tid];
    atomicAdd(&segSum[(size_t)lbl * DD + tid], v);
  }
  if (tid < ROWS) atomicAdd(&segCnt[sLab[tid]], 1);
}

// ------------------------------------------------------------- finalize -----
__global__ __launch_bounds__(256)
void dcn_finalize(const float* __restrict__ cen, const int* __restrict__ count,
                  const float* __restrict__ segSum, const int* __restrict__ segCnt,
                  const float* __restrict__ lossAcc, float* __restrict__ outLoss,
                  float* __restrict__ outCenters, float* __restrict__ outCount) {
  const int k = blockIdx.x;
  const int d = threadIdx.x;
  float cf  = (float)count[k];
  int   nci = count[k] + segCnt[k];
  float nf  = (float)nci;
  size_t o  = (size_t)k * DD + d;
  outCenters[o] = (cf * cen[o] + segSum[o]) / nf;
  if (d == 0) outCount[k] = (float)nci;
  if (k == 0 && d == 0) outLoss[0] = lossAcc[0] / (float)NN;
}

// ------------------------------------------------------------- launcher -----
extern "C" void kernel_launch(void* const* d_in, const int* in_sizes, int n_in,
                              void* d_out, int out_size, void* d_ws, size_t ws_size,
                              hipStream_t stream) {
  const float* emb   = (const float*)d_in[0];   // [N, D]
  const float* cen   = (const float*)d_in[1];   // [K, D]
  const int*   count = (const int*)d_in[2];     // [K]

  float* out        = (float*)d_out;
  float* outLoss    = out;                       // 1
  float* outLabels  = out + 1;                   // N
  float* outCenters = out + 1 + NN;              // K*D
  float* outCount   = out + 1 + NN + (size_t)KK * DD; // K

  float* ws      = (float*)d_ws;
  float* b2      = ws;                                 // K floats
  float* segSum  = ws + KK;                            // K*D floats
  int*   segCnt  = (int*)(ws + KK + (size_t)KK * DD);  // K ints
  float* lossAcc = ws + KK + (size_t)KK * DD + KK;     // 1 float
  unsigned short* cenBf =
      (unsigned short*)(ws + KK + (size_t)KK * DD + KK + 1); // K*D bf16

  const int nAcc = KK * DD + KK + 1;             // segSum + segCnt + loss
  dcn_zero<<<(nAcc + 255) / 256, 256, 0, stream>>>(segSum, nAcc);
  dcn_prep<<<KK / 8, 256, 0, stream>>>(cen, b2, cenBf);
  dcn_main<<<NN / ROWS, 256, 0, stream>>>(emb, cen, cenBf, b2, outLabels,
                                          segSum, segCnt, lossAcc);
  dcn_finalize<<<KK, 256, 0, stream>>>(cen, count, segSum, segCnt, lossAcc,
                                       outLoss, outCenters, outCount);
}
